// Net_5059471475239
// MI455X (gfx1250) — matrix operand order repaired and checked by reading, not compiled, not run on hardware
//
#include <hip/hip_runtime.h>

#define NN 4000
#define NE 48000
#define Y00f 0.28209479177387814f

typedef __attribute__((ext_vector_type(16))) __bf16 v16bf;
typedef __attribute__((ext_vector_type(8)))  float  v8f;

struct RadialP { const float *W1,*b1,*g1,*be1,*W2,*b2,*g2,*be2,*W3,*b3; };

__device__ __forceinline__ unsigned encf(float x){
  unsigned u = __float_as_uint(x);
  return (u & 0x80000000u) ? ~u : (u | 0x80000000u);
}
__device__ __forceinline__ float decf(unsigned e){
  return (e & 0x80000000u) ? __uint_as_float(e & 0x7FFFFFFFu) : __uint_as_float(~e);
}

__device__ __forceinline__ void ln_relu32(float* x, const float* g, const float* b){
  float mu = 0.f;
  #pragma unroll
  for (int j=0;j<32;++j) mu += x[j];
  mu *= (1.f/32.f);
  float var = 0.f;
  #pragma unroll
  for (int j=0;j<32;++j){ float d = x[j]-mu; var += d*d; }
  var *= (1.f/32.f);
  float rs = rsqrtf(var + 1e-5f);
  #pragma unroll
  for (int j=0;j<32;++j) x[j] = fmaxf((x[j]-mu)*rs*g[j] + b[j], 0.f);
}

__device__ __forceinline__ void radial2(const float* feat, const RadialP p, float* out){
  float h[32];
  #pragma unroll
  for (int j=0;j<32;++j){
    float a = p.b1[j];
    #pragma unroll
    for (int c=0;c<13;++c) a = fmaf(p.W1[j*13+c], feat[c], a);
    h[j] = a;
  }
  ln_relu32(h, p.g1, p.be1);
  #pragma unroll
  for (int j=0;j<32;++j){
    float a = p.b2[j];
    #pragma unroll
    for (int c=0;c<32;++c) a = fmaf(p.W2[j*32+c], h[c], a);
    out[j] = a;
  }
  ln_relu32(out, p.g2, p.be2);
}

__device__ __forceinline__ void rad_apply48(const float* h2, const float* W3, const float* b3,
                                            const float* f0s, float* out8){
  #pragma unroll
  for (int o=0;o<8;++o){
    float acc = 0.f;
    #pragma unroll
    for (int i=0;i<6;++i){
      float r = b3[o*6+i];
      #pragma unroll
      for (int c=0;c<32;++c) r = fmaf(W3[(o*6+i)*32+c], h2[c], r);
      acc = fmaf(r, f0s[i], acc);
    }
    out8[o] = acc * Y00f;
  }
}

// ---------------- small utility kernels ----------------
__global__ void k_zero(float* __restrict__ p, int n){
  int i = blockIdx.x*blockDim.x + threadIdx.x;
  if (i < n) p[i] = 0.f;
}
__global__ void k_smaxinit(unsigned* __restrict__ p, int n){
  int i = blockIdx.x*blockDim.x + threadIdx.x;
  if (i < n) p[i] = 0x007FFFFFu;   // encf(-inf)
}
// W3c[4096,32] f32 + b3c[4096] -> Wb bf16, layout [o][i(0..32)][c]
__global__ void k_wbprep(const float* __restrict__ W3, const float* __restrict__ b3,
                         __bf16* __restrict__ Wb){
  int idx = blockIdx.x*blockDim.x + threadIdx.x;
  if (idx >= 128*33*32) return;
  int c = idx & 31;
  int rest = idx >> 5;
  int i = rest % 33;
  int o = rest / 33;
  float v = (i < 32) ? W3[(o*32 + i)*32 + c] : b3[o*32 + c];
  Wb[idx] = (__bf16)v;
}
// q = f0 @ Wq^T  : [N,8]
__global__ void k_q(const float* __restrict__ f, const float* __restrict__ Wq,
                    float* __restrict__ q){
  int n = blockIdx.x*blockDim.x + threadIdx.x;
  if (n >= NN) return;
  float f0[6];
  #pragma unroll
  for (int i=0;i<6;++i) f0[i] = f[n*6+i];
  #pragma unroll
  for (int o=0;o<8;++o){
    float a = 0.f;
    #pragma unroll
    for (int i=0;i<6;++i) a = fmaf(Wq[o*6+i], f0[i], a);
    q[n*8+o] = a;
  }
}

// ---------------- per-edge radial nets + attention logits ----------------
__global__ void k_edge(const float* __restrict__ f, const float* __restrict__ pos,
                       const float* __restrict__ w, const int* __restrict__ src,
                       const int* __restrict__ dst, RadialP rv, RadialP rk, RadialP rc,
                       const float* __restrict__ q, float* __restrict__ escore,
                       float* __restrict__ vout, float* __restrict__ h2c,
                       unsigned* __restrict__ segmax, float* __restrict__ deg){
  int e = blockIdx.x*blockDim.x + threadIdx.x;
  if (e >= NE) return;
  int s = src[e], d = dst[e];
  float feat[13];
  #pragma unroll
  for (int c=0;c<12;++c) feat[c] = w[e*12+c];
  float dx = pos[d*3+0]-pos[s*3+0];
  float dy = pos[d*3+1]-pos[s*3+1];
  float dz = pos[d*3+2]-pos[s*3+2];
  feat[12] = sqrtf(dx*dx + dy*dy + dz*dz + 1e-8f);
  float f0s[6];
  #pragma unroll
  for (int i=0;i<6;++i) f0s[i] = f[s*6+i];

  float h2[32], t8[8];
  radial2(feat, rv, h2);
  rad_apply48(h2, rv.W3, rv.b3, f0s, t8);
  #pragma unroll
  for (int o=0;o<8;++o) vout[e*8+o] = t8[o];

  radial2(feat, rk, h2);
  rad_apply48(h2, rk.W3, rk.b3, f0s, t8);
  float sc = 0.f;
  #pragma unroll
  for (int o=0;o<8;++o) sc = fmaf(t8[o], q[d*8+o], sc);
  sc *= 0.35355339059327373f;   // 1/sqrt(8)
  escore[e] = sc;
  atomicMax(&segmax[d], encf(sc));
  atomicAdd(&deg[d], 1.f);

  radial2(feat, rc, h2);
  #pragma unroll
  for (int c=0;c<32;++c) h2c[e*32+c] = h2[c];
}

__global__ void k_soft1(const float* __restrict__ escore, const int* __restrict__ dst,
                        const unsigned* __restrict__ smax, float* __restrict__ aexp,
                        float* __restrict__ ssum){
  int e = blockIdx.x*blockDim.x + threadIdx.x;
  if (e >= NE) return;
  int d = dst[e];
  float a = expf(escore[e] - decf(smax[d]));
  aexp[e] = a;
  atomicAdd(&ssum[d], a);
}

__global__ void k_soft2(const float* __restrict__ aexp, const int* __restrict__ dst,
                        const float* __restrict__ ssum, const float* __restrict__ v,
                        float* __restrict__ attn){
  int e = blockIdx.x*blockDim.x + threadIdx.x;
  if (e >= NE) return;
  int d = dst[e];
  float alpha = aexp[e] / ssum[d];
  #pragma unroll
  for (int o=0;o<8;++o) atomicAdd(&attn[d*8+o], alpha * v[e*8+o]);
}

// ---------------- node update: Wp projection + GNormSE3 ----------------
__global__ void k_node(const float* __restrict__ f, const float* __restrict__ attn,
                       const float* __restrict__ Wp, const float* __restrict__ ng,
                       const float* __restrict__ nb, const float* __restrict__ nW,
                       const float* __restrict__ nbb, float* __restrict__ h){
  int n = blockIdx.x*blockDim.x + threadIdx.x;
  if (n >= NN) return;
  float zin[14];
  #pragma unroll
  for (int o=0;o<8;++o) zin[o] = attn[n*8+o];
  #pragma unroll
  for (int i=0;i<6;++i) zin[8+i] = f[n*6+i];
  float z[32], t[32];
  #pragma unroll
  for (int j=0;j<32;++j){
    float a = 0.f;
    #pragma unroll
    for (int c=0;c<14;++c) a = fmaf(Wp[j*14+c], zin[c], a);
    z[j] = a;
    t[j] = fmaxf(fabsf(a), 1e-12f);
  }
  ln_relu32(t, ng, nb);
  #pragma unroll
  for (int j=0;j<32;++j){
    float a = nbb[j];
    #pragma unroll
    for (int c=0;c<32;++c) a = fmaf(nW[j*32+c], t[c], a);
    float nrm = fmaxf(fabsf(z[j]), 1e-12f);
    h[n*32+j] = a * (z[j] / nrm);
  }
}

// S = h @ Wself^T : [N,128]
__global__ void k_self(const float* __restrict__ h, const float* __restrict__ Wself,
                       float* __restrict__ S){
  int n = blockIdx.x;
  int o = threadIdx.x;
  __shared__ float hn[32];
  if (o < 32) hn[o] = h[n*32 + o];
  __syncthreads();
  float a = 0.f;
  #pragma unroll
  for (int c=0;c<32;++c) a = fmaf(Wself[o*32+c], hn[c], a);
  S[n*128 + o] = a;
}

// ---------------- the dominant GEMM: WMMA bf16, f32 accumulate ----------------
// block = 16 edges x 128 outputs. 8 waves, each owns one 16x16 N-tile.
// K = 33 steps of 32: step i<32: A[m,c] = h2[m,c] * (Y00*hs[m,i]); step 32: A = Y00*hs (bias).
__global__ void k_gemm(const float* __restrict__ h2c, const float* __restrict__ h,
                       const int* __restrict__ src, const int* __restrict__ dst,
                       const __bf16* __restrict__ Wb, float* __restrict__ accum){
  __shared__ float sh2[16][33];
  __shared__ float shs[16][33];
  __shared__ int   ssrc[16];
  __shared__ int   sdst[16];
  const int e0  = blockIdx.x * 16;
  const int tid = threadIdx.x;
  if (tid < 16){ ssrc[tid] = src[e0 + tid]; sdst[tid] = dst[e0 + tid]; }
  __syncthreads();
  for (int idx = tid; idx < 512; idx += 256){
    int m = idx >> 5, c = idx & 31;
    sh2[m][c] = h2c[(e0 + m)*32 + c];
    shs[m][c] = h[ssrc[m]*32 + c] * Y00f;
  }
  __syncthreads();

  const int wave = tid >> 5, lane = tid & 31;
  const int row  = lane & 15, kh = lane >> 4;

  // A fragment K-runs (contiguous, matching B): lanes 0-15 K=0..15, lanes 16-31 K=16..31
  float h2run[16], hsrun[16];
  #pragma unroll
  for (int j=0;j<8;++j){
    h2run[j]   = sh2[row][kh*16 + j];
    h2run[8+j] = sh2[row][kh*16 + 8 + j];
    hsrun[j]   = shs[row][kh*16 + j];
    hsrun[8+j] = shs[row][kh*16 + 8 + j];
  }

  const int o = wave*16 + row;                       // absolute output channel for this lane's B column
  const __bf16* wb = Wb + (size_t)o*33*32 + (size_t)kh*16;  // B: lanes 0-15 K=0..15, 16-31 K=16..31

  v8f acc = 0.0f;
  for (int i=0;i<32;++i){
    float s = shs[row][i];
    v16bf a;
    #pragma unroll
    for (int j=0;j<16;++j) a[j] = (__bf16)(h2run[j] * s);
    v16bf b = *reinterpret_cast<const v16bf*>(wb + (size_t)i*32);
    acc = __builtin_amdgcn_wmma_f32_16x16x32_bf16(false, a, false, b, (short)0, acc, false, false);
  }
  { // bias K-step
    v16bf a;
    #pragma unroll
    for (int j=0;j<16;++j) a[j] = (__bf16)hsrun[j];
    v16bf b = *reinterpret_cast<const v16bf*>(wb + (size_t)32*32);
    acc = __builtin_amdgcn_wmma_f32_16x16x32_bf16(false, a, false, b, (short)0, acc, false, false);
  }
  // D layout: VGPR r -> M=r (lanes 0-15) / M=r+8 (lanes 16-31), N = lane&15
  #pragma unroll
  for (int r=0;r<8;++r){
    int m = r + kh*8;
    atomicAdd(&accum[(size_t)sdst[m]*128 + o], acc[r]);
  }
}

__global__ void k_final(const float* __restrict__ acc, const float* __restrict__ S,
                        const float* __restrict__ deg, float* __restrict__ out){
  int idx = blockIdx.x*blockDim.x + threadIdx.x;
  if (idx >= NN*128) return;
  int n = idx >> 7;
  float dg = deg[n];
  float m = fmaxf(dg, 1.f);
  out[idx] = acc[idx] / m + (dg > 0.f ? S[idx] : 0.f);
}

extern "C" void kernel_launch(void* const* d_in, const int* in_sizes, int n_in,
                              void* d_out, int out_size, void* d_ws, size_t ws_size,
                              hipStream_t stream){
  (void)in_sizes; (void)n_in; (void)out_size; (void)ws_size;
  const float* f   = (const float*)d_in[0];
  const float* pos = (const float*)d_in[1];
  const float* w   = (const float*)d_in[2];
  const int*   src = (const int*)d_in[3];
  const int*   dst = (const int*)d_in[4];
  auto gp = [&](int i){ return (const float*)d_in[i]; };
  RadialP rv{gp(5),gp(6),gp(7),gp(8),gp(9),gp(10),gp(11),gp(12),gp(13),gp(14)};
  RadialP rk{gp(15),gp(16),gp(17),gp(18),gp(19),gp(20),gp(21),gp(22),gp(23),gp(24)};
  const float* Wq    = gp(25);
  const float* Wp    = gp(26);
  const float* ng    = gp(27);
  const float* nb    = gp(28);
  const float* nW    = gp(29);
  const float* nbb   = gp(30);
  RadialP rc{gp(31),gp(32),gp(33),gp(34),gp(35),gp(36),gp(37),gp(38),gp(39),gp(40)};
  const float* Wself = gp(41);

  float* ws = (float*)d_ws;
  float*    Q    = ws + 0;         // [N,8]
  float*    ESC  = ws + 32000;     // [E]
  unsigned* SMAX = (unsigned*)(ws + 80000);  // [N]
  float*    SSUM = ws + 84000;     // [N]
  float*    DEG  = ws + 88000;     // [N]
  float*    AEXP = ws + 92000;     // [E]
  float*    V    = ws + 140000;    // [E,8]
  float*    ATTN = ws + 524000;    // [N,8]
  float*    H    = ws + 556000;    // [N,32]
  float*    S    = ws + 684000;    // [N,128]
  float*    ACC  = ws + 1196000;   // [N,128]
  float*    H2C  = ws + 1708000;   // [E,32]
  __bf16*   WB   = (__bf16*)(ws + 3244000);  // [128*33*32] bf16
  float*    out  = (float*)d_out;

  k_zero    <<<(8000+255)/256,   256, 0, stream>>>(SSUM, 8000);      // ssum + deg (contiguous)
  k_zero    <<<(32000+255)/256,  256, 0, stream>>>(ATTN, 32000);
  k_zero    <<<(512000+255)/256, 256, 0, stream>>>(ACC, 512000);
  k_smaxinit<<<(4000+255)/256,   256, 0, stream>>>(SMAX, 4000);
  k_wbprep  <<<(128*33*32+255)/256, 256, 0, stream>>>(rc.W3, rc.b3, WB);
  k_q       <<<(NN+255)/256, 256, 0, stream>>>(f, Wq, Q);
  k_edge    <<<(NE+255)/256, 256, 0, stream>>>(f, pos, w, src, dst, rv, rk, rc,
                                               Q, ESC, V, H2C, SMAX, DEG);
  k_soft1   <<<(NE+255)/256, 256, 0, stream>>>(ESC, dst, SMAX, AEXP, SSUM);
  k_soft2   <<<(NE+255)/256, 256, 0, stream>>>(AEXP, dst, SSUM, V, ATTN);
  k_node    <<<(NN+255)/256, 256, 0, stream>>>(f, ATTN, Wp, ng, nb, nW, nbb, H);
  k_self    <<<NN, 128, 0, stream>>>(H, Wself, S);
  k_gemm    <<<NE/16, 256, 0, stream>>>(H2C, H, src, dst, WB, ACC);
  k_final   <<<(NN*128+255)/256, 256, 0, stream>>>(ACC, S, DEG, out);
  hipMemcpyAsync(out + 512000, pos, (size_t)NN*3*sizeof(float),
                 hipMemcpyDeviceToDevice, stream);
}
